// TransformerBlock_83605833384266
// MI455X (gfx1250) — matrix operand order repaired and checked by reading, not compiled, not run on hardware
//
#include <hip/hip_runtime.h>
#include <hip/hip_fp16.h>

#define DIM   256
#define SEQ   1024
#define BATCH 8
#define HEADS 8
#define HDIM  32
#define HID   1024
#define MROWS (BATCH*SEQ)   // 8192

typedef __attribute__((ext_vector_type(16))) _Float16 v16h;
typedef __attribute__((ext_vector_type(8)))  float    v8f;

union Frag16 { v16h v; uint4 q[2]; };

__device__ __forceinline__ v8f wmma_f16(const Frag16& a, const Frag16& b, v8f c) {
  return __builtin_amdgcn_wmma_f32_16x16x32_f16(false, a.v, false, b.v, (short)0, c, false, false);
}

// ---------------------------------------------------------------------------
// Generic WMMA GEMM:  C[M,N] = epi( A[M,K](f16) * B[N,K]^T(f16) + bias )
// block = 128 threads (4 waves), each wave: 16(M) x 64(N), K-step 32.
// ---------------------------------------------------------------------------
enum { EPI_QKV=0, EPI_OPROJ, EPI_GELU, EPI_MLP2, EPI_CONVT, EPI_H16, EPI_F32B, EPI_DEN };

struct GemmArgs {
  const __half* A; const __half* B; const float* bias;
  long aBatch, bBatch;           // per-blockIdx.z element strides
  int M, N, K;
  float*  outF;
  __half* outH; __half* outH2; __half* outH3;
  const float* resid; const float* add2;
};

template<int EPI>
__global__ void gemm_wmma_kernel(GemmArgs g) {
  const int lane = threadIdx.x & 31;
  const int wid  = threadIdx.x >> 5;
  const int idx  = lane & 15;
  const int hfb  = lane >> 4;
  const int z    = blockIdx.z;
  const __half* A = g.A + (long)z * g.aBatch;
  const __half* B = g.B + (long)z * g.bBatch;
  const int mRow = blockIdx.y * 64 + wid * 16;
  const int n0   = blockIdx.x * 64;
  const int K    = g.K;

  v8f acc[4];
  #pragma unroll
  for (int t = 0; t < 4; ++t)
    #pragma unroll
    for (int r = 0; r < 8; ++r) acc[t][r] = 0.f;

  const __half* pa0 = A + (long)(mRow + idx) * K + hfb * 16;    // A-fragment pattern
  for (int kb = 0; kb < K; kb += 32) {
    Frag16 af;
    { const __half* p = pa0 + kb;
      af.q[0] = *(const uint4*)p;          // K = kb+16*hfb+{0..7}
      af.q[1] = *(const uint4*)(p + 8); }  // K = kb+16*hfb+{8..15}
    #pragma unroll
    for (int t = 0; t < 4; ++t) {
      Frag16 bf;
      const __half* p = B + (long)(n0 + 16*t + idx) * K + kb + hfb * 16; // B: contiguous 16 per half
      bf.q[0] = *(const uint4*)p;
      bf.q[1] = *(const uint4*)(p + 8);
      acc[t] = wmma_f16(af, bf, acc[t]);
    }
  }

  #pragma unroll
  for (int t = 0; t < 4; ++t) {
    const int nt = n0 + 16 * t;
    #pragma unroll
    for (int r = 0; r < 8; ++r) {
      const int gm = mRow + r + 8 * hfb;   // C/D layout: row = r + 8*(lane>=16)
      const int gn = nt + idx;             //             col = lane&15
      float val = acc[t][r];
      if (g.bias) val += g.bias[gn];

      if (EPI == EPI_QKV) {
        const int b = gm >> 10, s = gm & 1023;
        const int sec = gn >> 8, rem = gn & 255;
        const int hd = rem >> 5, dd = rem & 31;
        const long bh = (long)(b * HEADS + hd);
        const __half hv = __float2half(val);
        if (sec == 0)      g.outH [(bh * SEQ + s) * HDIM + dd] = hv;  // Q [b,h,s,d]
        else if (sec == 1) g.outH2[(bh * SEQ + s) * HDIM + dd] = hv;  // K [b,h,s,d]
        else               g.outH3[(bh * HDIM + dd) * SEQ + s] = hv;  // V^T [b,h,d,s]
      } else if (EPI == EPI_OPROJ) {
        const long o = (long)gm * g.N + gn;
        g.outF[o] = val + g.resid[o];                                 // t2 = t + o@Wo^T+bo
      } else if (EPI == EPI_GELU) {
        const float gv = 0.5f * val * (1.f + erff(val * 0.70710678118f));
        g.outH[(long)gm * g.N + gn] = __float2half(gv);
      } else if (EPI == EPI_MLP2) {
        const int b = gm >> 10, s = gm & 1023;
        const long o = (long)gm * DIM + gn;
        const float v2 = val + g.resid[o];                            // x2 = t2 + mlp
        g.outH[o] = __float2half(v2);                                 // x2 f16 [b,s,c]
        g.outF[((long)(b * DIM + gn)) * SEQ + s] = v2;                // x2 f32 [b,c,s]
      } else if (EPI == EPI_CONVT) {
        const int b = gm >> 10, s = gm & 1023;
        g.outH[((long)(b * DIM + gn)) * SEQ + s] = __float2half(val); // [b,c,s]
      } else if (EPI == EPI_H16) {
        g.outH[(long)gm * g.N + gn] = __float2half(val);              // [b,s,c]
      } else if (EPI == EPI_F32B) {
        g.outF[((long)z * g.M + gm) * g.N + gn] = val;                // channel logits
      } else if (EPI == EPI_DEN) {
        const long o = ((long)(z * DIM + gn)) * SEQ + gm;             // gm = s, gn = c
        g.outF[o] = g.resid[o] + 0.1f * val + 0.1f * g.add2[o];       // x2 + .1 den + .1 local
      }
    }
  }
}

// ---------------------------------------------------------------------------
// Flash-style attention. grid (16, HEADS, BATCH), block 128 (4 waves).
// Each wave owns 16 query rows; streams keys in tiles of 32 with online softmax.
// ---------------------------------------------------------------------------
__global__ void attn_kernel(const __half* qp, const __half* kp, const __half* vTp, __half* op) {
  const int lane = threadIdx.x & 31;
  const int wid  = threadIdx.x >> 5;
  const int idx  = lane & 15;
  const int hfb  = lane >> 4;
  const int b = blockIdx.z, h = blockIdx.y;
  const int qRow0 = blockIdx.x * 64 + wid * 16;
  const long bh = (long)(b * HEADS + h);

  __shared__ __half P[4][16][32];

  Frag16 qa;  // Q as A-matrix 16x32 (K = head dim)
  { const __half* p = qp + (bh * SEQ + qRow0 + idx) * HDIM + hfb * 16;
    qa.q[0] = *(const uint4*)p; qa.q[1] = *(const uint4*)(p + 8); }

  v8f o0, o1;
  float mi[8], li[8];
  #pragma unroll
  for (int r = 0; r < 8; ++r) { o0[r] = 0.f; o1[r] = 0.f; mi[r] = -1e30f; li[r] = 0.f; }

  const float scale = 0.17677669529f;   // 1/sqrt(32)

  for (int kt = 0; kt < SEQ; kt += 32) {
    Frag16 kb0, kb1;  // K^T as B-matrix: col = key, K = head dim (contiguous 16 per half)
    { const __half* p = kp + (bh * SEQ + kt + idx) * HDIM + hfb * 16;
      kb0.q[0] = *(const uint4*)p; kb0.q[1] = *(const uint4*)(p + 8); }
    { const __half* p = kp + (bh * SEQ + kt + 16 + idx) * HDIM + hfb * 16;
      kb1.q[0] = *(const uint4*)p; kb1.q[1] = *(const uint4*)(p + 8); }

    v8f s0, s1;
    #pragma unroll
    for (int r = 0; r < 8; ++r) { s0[r] = 0.f; s1[r] = 0.f; }
    s0 = wmma_f16(qa, kb0, s0);
    s1 = wmma_f16(qa, kb1, s1);

    float ps0[8], ps1[8];
    #pragma unroll
    for (int r = 0; r < 8; ++r) {
      const float a0 = s0[r] * scale, a1 = s1[r] * scale;
      float mx = fmaxf(a0, a1);
      #pragma unroll
      for (int off = 1; off < 16; off <<= 1) mx = fmaxf(mx, __shfl_xor(mx, off, 32));
      const float nm = fmaxf(mi[r], mx);
      const float rsc = __expf(mi[r] - nm);
      o0[r] *= rsc; o1[r] *= rsc; li[r] *= rsc; mi[r] = nm;
      const float p0 = __expf(a0 - nm), p1 = __expf(a1 - nm);
      float sm = p0 + p1;
      #pragma unroll
      for (int off = 1; off < 16; off <<= 1) sm += __shfl_xor(sm, off, 32);
      li[r] += sm;
      ps0[r] = p0; ps1[r] = p1;
    }

    // D-layout probs -> LDS -> A-layout fragment
    #pragma unroll
    for (int r = 0; r < 8; ++r) {
      const int m = r + 8 * hfb;
      P[wid][m][idx]      = __float2half(ps0[r]);
      P[wid][m][idx + 16] = __float2half(ps1[r]);
    }
    __syncthreads();

    Frag16 pf;
    { const __half* p = &P[wid][idx][hfb * 16];
      pf.q[0] = *(const uint4*)p; pf.q[1] = *(const uint4*)(p + 8); }

    Frag16 v0, v1;  // V as B-matrix: col = d, K = key (contiguous in V^T)
    { const __half* p = vTp + (bh * HDIM + idx) * SEQ + kt + hfb * 16;
      v0.q[0] = *(const uint4*)p; v0.q[1] = *(const uint4*)(p + 8); }
    { const __half* p = vTp + (bh * HDIM + idx + 16) * SEQ + kt + hfb * 16;
      v1.q[0] = *(const uint4*)p; v1.q[1] = *(const uint4*)(p + 8); }

    o0 = wmma_f16(pf, v0, o0);
    o1 = wmma_f16(pf, v1, o1);
  }

  #pragma unroll
  for (int r = 0; r < 8; ++r) {
    const int m = r + 8 * hfb;
    const float inv = 1.0f / li[r];
    const long base = ((long)(b * SEQ + qRow0 + m)) * DIM + h * HDIM;
    op[base + idx]      = __float2half(o0[r] * inv);
    op[base + 16 + idx] = __float2half(o1[r] * inv);
  }
}

// ---------------------------------------------------------------------------
// LayerNorm over 256 channels; one wave per token row (8 waves / block).
// transposed: read x[b,c,s] (NCHW gather); tOut optionally keeps f32 copy.
// ---------------------------------------------------------------------------
__global__ void ln_kernel(const float* in, const float* gam, const float* bet,
                          float* tOut, __half* nOut, int transposed) {
  const int lane = threadIdx.x & 31;
  const int row  = blockIdx.x * 8 + (threadIdx.x >> 5);
  const int b = row >> 10, s = row & 1023;
  float v[8], sum = 0.f, sq = 0.f;
  #pragma unroll
  for (int j = 0; j < 8; ++j) {
    const int c = j * 32 + lane;
    const float x = transposed ? in[((long)(b * DIM + c)) * SEQ + s]
                               : in[(long)row * DIM + c];
    v[j] = x; sum += x; sq += x * x;
  }
  #pragma unroll
  for (int off = 1; off < 32; off <<= 1) {
    sum += __shfl_xor(sum, off, 32);
    sq  += __shfl_xor(sq,  off, 32);
  }
  const float mu  = sum * (1.f / DIM);
  const float var = sq * (1.f / DIM) - mu * mu;
  const float rs  = rsqrtf(var + 1e-5f);
  #pragma unroll
  for (int j = 0; j < 8; ++j) {
    const int c = j * 32 + lane;
    if (tOut) tOut[(long)row * DIM + c] = v[j];
    nOut[(long)row * DIM + c] = __float2half((v[j] - mu) * rs * gam[c] + bet[c]);
  }
}

// Channel-attention softmax over 256 logits; one wave per (b,c) row.
__global__ void chan_softmax_kernel(const float* aF, __half* aH) {
  const int lane = threadIdx.x & 31;
  const int row  = blockIdx.x * 8 + (threadIdx.x >> 5);   // 2048 rows
  float v[8], mx = -1e30f;
  #pragma unroll
  for (int j = 0; j < 8; ++j) { v[j] = aF[(long)row * DIM + j * 32 + lane]; mx = fmaxf(mx, v[j]); }
  #pragma unroll
  for (int off = 1; off < 32; off <<= 1) mx = fmaxf(mx, __shfl_xor(mx, off, 32));
  float sm = 0.f;
  #pragma unroll
  for (int j = 0; j < 8; ++j) { v[j] = __expf(v[j] - mx); sm += v[j]; }
  #pragma unroll
  for (int off = 1; off < 32; off <<= 1) sm += __shfl_xor(sm, off, 32);
  const float inv = 1.0f / sm;
  #pragma unroll
  for (int j = 0; j < 8; ++j)
    aH[(long)row * DIM + j * 32 + lane] = __float2half(v[j] * inv);
}

// Depthwise 3x3 (SAME) on x2[b,c,32,32]; trivial FLOPs, one thread per element.
__global__ void dwconv_kernel(const float* x2, const float* w, const float* bias, float* out) {
  const int i = blockIdx.x * 256 + threadIdx.x;           // 2^21 elements
  const int b = i >> 18, c = (i >> 10) & 255, s = i & 1023;
  const int hh = s >> 5, ww = s & 31;
  const long base = ((long)(b * DIM + c)) << 10;
  float acc = bias[c];
  #pragma unroll
  for (int dy = -1; dy <= 1; ++dy) {
    const int y = hh + dy;
    if ((unsigned)y >= 32u) continue;
    #pragma unroll
    for (int dx = -1; dx <= 1; ++dx) {
      const int x = ww + dx;
      if ((unsigned)x >= 32u) continue;
      acc += x2[base + (y << 5) + x] * w[c * 9 + (dy + 1) * 3 + (dx + 1)];
    }
  }
  out[i] = acc;
}

__global__ void cvt_f16_kernel(const float* in, __half* out, int n) {
  const int i = blockIdx.x * 256 + threadIdx.x;
  if (i < n) out[i] = __float2half(in[i]);
}

// ---------------------------------------------------------------------------
extern "C" void kernel_launch(void* const* d_in, const int* in_sizes, int n_in,
                              void* d_out, int out_size, void* d_ws, size_t ws_size,
                              hipStream_t stream) {
  const float* x    = (const float*)d_in[0];
  const float* ln1g = (const float*)d_in[1];
  const float* ln1b = (const float*)d_in[2];
  const float* wqkv = (const float*)d_in[3];
  const float* bqkv = (const float*)d_in[4];
  const float* wo   = (const float*)d_in[5];
  const float* bo   = (const float*)d_in[6];
  const float* ln2g = (const float*)d_in[7];
  const float* ln2b = (const float*)d_in[8];
  const float* w1   = (const float*)d_in[9];
  const float* b1   = (const float*)d_in[10];
  const float* w2   = (const float*)d_in[11];
  const float* b2   = (const float*)d_in[12];
  const float* c1w  = (const float*)d_in[13];
  const float* c1b  = (const float*)d_in[14];
  const float* c2w  = (const float*)d_in[15];
  const float* c2b  = (const float*)d_in[16];
  const float* c3w  = (const float*)d_in[17];
  const float* c3b  = (const float*)d_in[18];
  const float* dww  = (const float*)d_in[19];
  const float* dwb  = (const float*)d_in[20];
  float* out = (float*)d_out;

  char* wsp = (char*)d_ws;
  size_t off = 0;
  auto carve = [&](size_t bytes) -> void* {
    void* p = wsp + off; off = (off + bytes + 255) & ~(size_t)255; return p;
  };
  float*  t_f   = (float*) carve((size_t)MROWS * DIM * 4);    // t (pre-LN residual)
  __half* tn_h  = (__half*)carve((size_t)MROWS * DIM * 2);    // LN1(t)
  __half* q_h   = (__half*)carve((size_t)MROWS * DIM * 2);    // [b,h,s,d]
  __half* k_h   = (__half*)carve((size_t)MROWS * DIM * 2);    // [b,h,s,d]
  __half* vT_h  = (__half*)carve((size_t)MROWS * DIM * 2);    // [b,h,d,s]
  __half* o_h   = (__half*)carve((size_t)MROWS * DIM * 2);    // attn out [b,s,c]
  float*  t2_f  = (float*) carve((size_t)MROWS * DIM * 4);    // t + attn
  __half* tn2_h = (__half*)carve((size_t)MROWS * DIM * 2);    // LN2
  __half* hdn_h = (__half*)carve((size_t)MROWS * HID * 2);    // gelu(mlp1)
  __half* x2_h  = (__half*)carve((size_t)MROWS * DIM * 2);    // x2 [b,s,c]
  float*  x2_f  = (float*) carve((size_t)MROWS * DIM * 4);    // x2 [b,c,s]
  __half* qf_h  = (__half*)carve((size_t)MROWS * DIM * 2);    // [b,c,s]
  __half* kf_h  = (__half*)carve((size_t)MROWS * DIM * 2);    // [b,c,s]
  __half* vf_h  = (__half*)carve((size_t)MROWS * DIM * 2);    // [b,s,c]
  float*  a_f   = (float*) carve((size_t)BATCH * DIM * DIM * 4);
  __half* a_h   = (__half*)carve((size_t)BATCH * DIM * DIM * 2);
  float*  loc_f = (float*) carve((size_t)MROWS * DIM * 4);
  __half* wqkv_h= (__half*)carve((size_t)3 * DIM * DIM * 2);
  __half* wo_h  = (__half*)carve((size_t)DIM * DIM * 2);
  __half* w1_h  = (__half*)carve((size_t)HID * DIM * 2);
  __half* w2_h  = (__half*)carve((size_t)DIM * HID * 2);
  __half* c1_h  = (__half*)carve((size_t)DIM * DIM * 2);
  __half* c2_h  = (__half*)carve((size_t)DIM * DIM * 2);
  __half* c3_h  = (__half*)carve((size_t)DIM * DIM * 2);

  auto cvt = [&](const float* src, __half* dst, int n) {
    cvt_f16_kernel<<<(n + 255) / 256, 256, 0, stream>>>(src, dst, n);
  };
  cvt(wqkv, wqkv_h, 3 * DIM * DIM);
  cvt(wo,   wo_h,   DIM * DIM);
  cvt(w1,   w1_h,   HID * DIM);
  cvt(w2,   w2_h,   DIM * HID);
  cvt(c1w,  c1_h,   DIM * DIM);
  cvt(c2w,  c2_h,   DIM * DIM);
  cvt(c3w,  c3_h,   DIM * DIM);

  // LN1 with NCHW->NSC transpose; keeps f32 residual copy
  ln_kernel<<<MROWS / 8, 256, 0, stream>>>(x, ln1g, ln1b, t_f, tn_h, 1);

  GemmArgs g{};
  auto launch = [&](int epi, dim3 grid) {
    switch (epi) {
      case EPI_QKV:   gemm_wmma_kernel<EPI_QKV>  <<<grid, 128, 0, stream>>>(g); break;
      case EPI_OPROJ: gemm_wmma_kernel<EPI_OPROJ><<<grid, 128, 0, stream>>>(g); break;
      case EPI_GELU:  gemm_wmma_kernel<EPI_GELU> <<<grid, 128, 0, stream>>>(g); break;
      case EPI_MLP2:  gemm_wmma_kernel<EPI_MLP2> <<<grid, 128, 0, stream>>>(g); break;
      case EPI_CONVT: gemm_wmma_kernel<EPI_CONVT><<<grid, 128, 0, stream>>>(g); break;
      case EPI_H16:   gemm_wmma_kernel<EPI_H16>  <<<grid, 128, 0, stream>>>(g); break;
      case EPI_F32B:  gemm_wmma_kernel<EPI_F32B> <<<grid, 128, 0, stream>>>(g); break;
      case EPI_DEN:   gemm_wmma_kernel<EPI_DEN>  <<<grid, 128, 0, stream>>>(g); break;
    }
  };

  // QKV projection
  g = GemmArgs{}; g.A = tn_h; g.B = wqkv_h; g.bias = bqkv;
  g.M = MROWS; g.N = 3 * DIM; g.K = DIM;
  g.outH = q_h; g.outH2 = k_h; g.outH3 = vT_h;
  launch(EPI_QKV, dim3(12, 128, 1));

  // Attention
  attn_kernel<<<dim3(16, HEADS, BATCH), 128, 0, stream>>>(q_h, k_h, vT_h, o_h);

  // Output projection + residual
  g = GemmArgs{}; g.A = o_h; g.B = wo_h; g.bias = bo;
  g.M = MROWS; g.N = DIM; g.K = DIM; g.outF = t2_f; g.resid = t_f;
  launch(EPI_OPROJ, dim3(4, 128, 1));

  // LN2
  ln_kernel<<<MROWS / 8, 256, 0, stream>>>(t2_f, ln2g, ln2b, nullptr, tn2_h, 0);

  // MLP1 + GELU
  g = GemmArgs{}; g.A = tn2_h; g.B = w1_h; g.bias = b1;
  g.M = MROWS; g.N = HID; g.K = DIM; g.outH = hdn_h;
  launch(EPI_GELU, dim3(16, 128, 1));

  // MLP2 + residual -> x2 in both layouts
  g = GemmArgs{}; g.A = hdn_h; g.B = w2_h; g.bias = b2;
  g.M = MROWS; g.N = DIM; g.K = HID; g.outH = x2_h; g.outF = x2_f; g.resid = t2_f;
  launch(EPI_MLP2, dim3(4, 128, 1));

  // 1x1 convs: qf,kf stored [b,c,s]; vf stored [b,s,c]
  g = GemmArgs{}; g.A = x2_h; g.B = c1_h; g.bias = c1b;
  g.M = MROWS; g.N = DIM; g.K = DIM; g.outH = qf_h;
  launch(EPI_CONVT, dim3(4, 128, 1));
  g.B = c2_h; g.bias = c2b; g.outH = kf_h;
  launch(EPI_CONVT, dim3(4, 128, 1));
  g.B = c3_h; g.bias = c3b; g.outH = vf_h;
  launch(EPI_H16, dim3(4, 128, 1));

  // Channel logits a[b,c,d] = qf . kf  (K = 1024 spatial), batched over b
  g = GemmArgs{}; g.A = qf_h; g.B = kf_h; g.bias = nullptr;
  g.aBatch = (long)DIM * SEQ; g.bBatch = (long)DIM * SEQ;
  g.M = DIM; g.N = DIM; g.K = SEQ; g.outF = a_f;
  launch(EPI_F32B, dim3(4, 4, BATCH));

  chan_softmax_kernel<<<(BATCH * DIM) / 8, 256, 0, stream>>>(a_f, a_h);

  // Depthwise 3x3
  dwconv_kernel<<<(MROWS * DIM) / 256, 256, 0, stream>>>(x2_f, dww, dwb, loc_f);

  // den = a @ vf, fused final combine: out = x2 + 0.1*den + 0.1*local
  g = GemmArgs{}; g.A = vf_h; g.B = a_h; g.bias = nullptr;
  g.aBatch = (long)SEQ * DIM; g.bBatch = (long)DIM * DIM;
  g.M = SEQ; g.N = DIM; g.K = DIM;
  g.outF = out; g.resid = x2_f; g.add2 = loc_f;
  launch(EPI_DEN, dim3(4, 16, BATCH));
}
